// DynamicConv1d_34746285425144
// MI455X (gfx1250) — matrix operand order repaired and checked by reading, not compiled, not run on hardware
//
#include <hip/hip_runtime.h>

typedef __attribute__((ext_vector_type(16))) __bf16 v16bf;
typedef __attribute__((ext_vector_type(8)))  __bf16 v8bf;
typedef __attribute__((ext_vector_type(8)))  float  v8f;
typedef __attribute__((ext_vector_type(4)))  unsigned int v4u;
typedef __attribute__((ext_vector_type(4)))  int v4i;
typedef __attribute__((ext_vector_type(8)))  int v8i;

#if defined(__has_builtin)
# if __has_builtin(__builtin_amdgcn_tensor_load_to_lds)
#  define HAVE_TDM 1
# else
#  define HAVE_TDM 0
# endif
#else
# define HAVE_TDM 0
#endif

namespace {
constexpr int B_N   = 32;
constexpr int C_IN  = 128;
constexpr int T_LEN = 4096;
constexpr int K_BR  = 4;
constexpr int C_OUT = 256;
constexpr int KS    = 7;
constexpr int PADL  = 3;
constexpr int T_PAD = T_LEN + 16;      // 8 zero rows each side in xbf

// workspace byte offsets
constexpr size_t WS_ALPHA  = 0;                       // B_N*K_BR floats
constexpr size_t WS_POOLED = 1024;                    // B_N*C_IN floats
constexpr size_t WS_WEFF   = 32768;                   // bf16 [B][C_OUT][KS][C_IN]
constexpr size_t WS_XBF    = 16ull * 1024 * 1024;     // bf16 [B][T_PAD][C_IN]
constexpr size_t XBF_BYTES = (size_t)B_N * T_PAD * C_IN * 2;
}

// ---------------- routing: mean pool over T ----------------
__global__ void pool_kernel(const float* __restrict__ x, float* __restrict__ pooled) {
  __shared__ float red[256];
  const int row = blockIdx.x;                    // b*C_IN + cin
  const float* xr = x + (size_t)row * T_LEN;
  float s = 0.f;
  for (int t = threadIdx.x; t < T_LEN; t += 256) s += xr[t];
  red[threadIdx.x] = s;
  __syncthreads();
  for (int off = 128; off > 0; off >>= 1) {
    if (threadIdx.x < off) red[threadIdx.x] += red[threadIdx.x + off];
    __syncthreads();
  }
  if (threadIdx.x == 0) pooled[row] = red[0] * (1.0f / T_LEN);
}

// ---------------- routing: linear + softmax over K=4 ----------------
__global__ void route_kernel(const float* __restrict__ pooled,
                             const float* __restrict__ rw,
                             const float* __restrict__ rb,
                             float* __restrict__ alpha) {
  const int b = threadIdx.x;
  if (b >= B_N) return;
  float lg[K_BR];
  float mx = -1e30f;
  #pragma unroll
  for (int k = 0; k < K_BR; ++k) {
    float acc = rb[k];
    const float* p = pooled + b * C_IN;
    const float* w = rw + k * C_IN;
    for (int c = 0; c < C_IN; ++c) acc += p[c] * w[c];
    lg[k] = acc;
    mx = fmaxf(mx, acc);
  }
  float den = 0.f;
  #pragma unroll
  for (int k = 0; k < K_BR; ++k) { lg[k] = __expf(lg[k] - mx); den += lg[k]; }
  #pragma unroll
  for (int k = 0; k < K_BR; ++k) alpha[b * K_BR + k] = lg[k] / den;
}

// ------- fold alpha into weights: w_eff[b][cout][s][cin] (bf16) -------
__global__ void weff_kernel(const float* __restrict__ weight,
                            const float* __restrict__ alpha,
                            __bf16* __restrict__ weff) {
  const int bid  = blockIdx.x;                   // b*C_OUT + cout
  const int b    = bid >> 8;
  const int cout = bid & (C_OUT - 1);
  const int ci   = threadIdx.x;                  // 0..127
  const float a0 = alpha[b*4+0], a1 = alpha[b*4+1];
  const float a2 = alpha[b*4+2], a3 = alpha[b*4+3];
  const size_t kstride = (size_t)C_OUT * C_IN * KS;
  #pragma unroll
  for (int s = 0; s < KS; ++s) {
    const size_t widx = ((size_t)cout * C_IN + ci) * KS + s;
    float v = a0 * weight[widx]
            + a1 * weight[widx + kstride]
            + a2 * weight[widx + 2*kstride]
            + a3 * weight[widx + 3*kstride];
    weff[(((size_t)(b * C_OUT + cout) * KS) + s) * C_IN + ci] = (__bf16)v;
  }
}

// ------- zero the halo pad rows of xbf (t in [0,8) and [T+8,T+16)) -------
__global__ void xpad_kernel(__bf16* __restrict__ xbf) {
  const int i  = blockIdx.x * 256 + threadIdx.x;  // 32*16*128 = 65536 total
  const int b  = i >> 11;
  const int r  = (i >> 7) & 15;
  const int ci = i & 127;
  const int t  = (r < 8) ? r : (T_LEN + 8 + (r - 8));
  xbf[((size_t)b * T_PAD + t) * C_IN + ci] = (__bf16)0.f;
}

// ------- transpose+convert x: f32 [b][cin][t] -> bf16 [b][t+8][cin] -------
__global__ void xbf_kernel(const float* __restrict__ x, __bf16* __restrict__ xbf) {
  __shared__ __bf16 tile[128 * 34];              // [cin][t], padded rows
  const int b  = blockIdx.y;
  const int t0 = blockIdx.x * 32;
  const int tid = threadIdx.x;
  const int tl = tid & 31;
  #pragma unroll
  for (int it = 0; it < 16; ++it) {
    const int ci = (tid >> 5) + it * 8;
    tile[ci * 34 + tl] = (__bf16)x[((size_t)b * C_IN + ci) * T_LEN + t0 + tl];
  }
  __syncthreads();
  #pragma unroll
  for (int it = 0; it < 16; ++it) {
    const int idx = tid + it * 256;
    const int tr  = idx >> 7;
    const int ci  = idx & 127;
    xbf[((size_t)b * T_PAD + t0 + tr + 8) * C_IN + ci] = tile[ci * 34 + tr];
  }
}

// ---------------- TDM helpers ----------------
__device__ __forceinline__ void wait_tensorcnt0() {
#if defined(__has_builtin) && __has_builtin(__builtin_amdgcn_s_wait_tensorcnt)
  __builtin_amdgcn_s_wait_tensorcnt(0);
#else
  asm volatile("s_wait_tensorcnt 0x0" ::: "memory");
#endif
  asm volatile("" ::: "memory");   // LDS written behind the compiler's back
}

#if HAVE_TDM
// 2D bf16 tile load: tile_x elems contiguous (dim0), tile_y rows, row stride
// in elements; LDS rows padded by pad_amount dwords every pad_interval dwords.
__device__ __forceinline__ void tdm_load_tile_bf16(
    unsigned lds_byte, const __bf16* gsrc,
    unsigned tile_x, unsigned tile_y, unsigned row_stride_elems,
    unsigned pad_interval_code, unsigned pad_amount_code) {
  const unsigned long long ga = (unsigned long long)(size_t)(const void*)gsrc;
  v4u g0;
  g0.x = 1u;                                                  // count=1, user D#
  g0.y = lds_byte;                                            // lds_addr
  g0.z = (unsigned)ga;                                        // global_addr lo
  g0.w = (unsigned)((ga >> 32) & 0x01ffffffu) | 0x80000000u;  // addr hi | type=2
  const unsigned td0 = 0x40000000u, td1 = 0x40000000u;        // huge dims: no OOB clip
  v8i g1;
  g1[0] = (int)((1u << 16)                                    // data_size = 2 B
              | (1u << 20)                                    // pad_enable
              | (pad_interval_code << 22)
              | (pad_amount_code << 25));
  g1[1] = (int)((td0 & 0xffffu) << 16);                       // dim0 lo16 (atomic addr = 0)
  g1[2] = (int)(((td0 >> 16) & 0xffffu) | ((td1 & 0xffffu) << 16));
  g1[3] = (int)(((td1 >> 16) & 0xffffu) | (tile_x << 16));    // dim1 hi16 | tile_dim0
  g1[4] = (int)(tile_y & 0xffffu);                            // tile_dim1, tile_dim2=0
  g1[5] = (int)row_stride_elems;                              // tensor_dim0_stride lo32
  g1[6] = 0;
  g1[7] = 0;
  v4i g2 = {0, 0, 0, 0};
  v4i g3 = {0, 0, 0, 0};
# if __clang_major__ >= 23
  v8i g4 = {0, 0, 0, 0, 0, 0, 0, 0};
  __builtin_amdgcn_tensor_load_to_lds(g0, g1, g2, g3, g4, 0);
# else
  __builtin_amdgcn_tensor_load_to_lds(g0, g1, g2, g3, 0);
# endif
}
#endif

// ---------------- main conv as 7-tap shifted WMMA GEMM ----------------
// MODE 0: stage from f32 x (convert in-kernel)        [ws fallback]
// MODE 1: stage bf16 from pre-transposed xbf (vector copies)
// MODE 2: stage bf16 from xbf via TDM, double-buffered
template <int MODE>
__global__ __launch_bounds__(256)
void conv_wmma_kernel(const float* __restrict__ x,
                      const __bf16* __restrict__ xbf,
                      const __bf16* __restrict__ weff,
                      const float* __restrict__ bias,
                      const float* __restrict__ alpha,
                      float* __restrict__ out) {
  constexpr int TT    = 128;
  constexpr int CT    = 128;
  constexpr int THALO = TT + KS - 1;  // 134
  constexpr int LROW  = 40;           // padded LDS row: 64B data + 16B pad
  constexpr int BUFSZ = THALO * LROW;

  __shared__ __attribute__((aligned(16))) __bf16 xs[2 * BUFSZ];

  const int tid  = threadIdx.x;
  const int lane = tid & 31;
  const int wave = tid >> 5;
  const int half = lane >> 4;
  const int lq   = lane & 15;

  const int b     = blockIdx.z;
  const int tTile = blockIdx.x * TT;
  const int cTile = blockIdx.y * CT;
  const int waveC = cTile + (wave & 3) * 32;     // 2 m-frags of 16
  const int wTloc = (wave >> 2) * 64;            // 4 n-frags of 16 (local)
  const int waveT = tTile + wTloc;

  v8f acc[2][4] = {};

#if HAVE_TDM
  if constexpr (MODE == 2) {
    const __bf16* xrow = xbf + ((size_t)(b * T_PAD + tTile + 5)) * C_IN;
    if (wave == 0) {
      tdm_load_tile_bf16((unsigned)(size_t)(void*)&xs[0], xrow,
                         32u, (unsigned)THALO, (unsigned)C_IN, 3u, 3u);
      wait_tensorcnt0();
    }
    __syncthreads();
  }
#endif

  for (int kc = 0; kc < C_IN / 32; ++kc) {
    const __bf16* xcur = xs;
    if constexpr (MODE == 2) {
      xcur = xs + (kc & 1) * BUFSZ;
#if HAVE_TDM
      const __bf16* xrow = xbf + ((size_t)(b * T_PAD + tTile + 5)) * C_IN;
      if (kc < 3 && wave == 0) {
        tdm_load_tile_bf16((unsigned)(size_t)(void*)&xs[((kc + 1) & 1) * BUFSZ],
                           xrow + (kc + 1) * 32,
                           32u, (unsigned)THALO, (unsigned)C_IN, 3u, 3u);
      }
#endif
    } else if constexpr (MODE == 1) {
      const __bf16* src = xbf + ((size_t)(b * T_PAD + tTile + 5)) * C_IN + kc * 32;
      #pragma unroll
      for (int i = 0; i < 3; ++i) {
        const int idx = tid + i * 256;           // one v8bf (16B) per slot
        const int tl  = idx >> 2;
        const int cq  = idx & 3;
        if (tl < THALO) {
          v8bf v = *(const v8bf*)(src + (size_t)tl * C_IN + cq * 8);
          *(v8bf*)&xs[tl * LROW + cq * 8] = v;
        }
      }
      __syncthreads();
    } else {
      const float* xb = x + ((size_t)b * C_IN + kc * 32) * T_LEN;
      #pragma unroll
      for (int p = 0; p < 4; ++p) {
        const int ci = (tid >> 5) + p * 8;
        #pragma unroll
        for (int q = 0; q < 5; ++q) {
          const int tl = (tid & 31) + q * 32;
          if (tl < THALO) {
            const int tg = tTile + tl - PADL;
            float v = (tg >= 0 && tg < T_LEN) ? xb[(size_t)ci * T_LEN + tg] : 0.f;
            xs[tl * LROW + ci] = (__bf16)v;
          }
        }
      }
      __syncthreads();
    }

    // ---- accumulate 7 taps x (Cin chunk of 32) ----
    const __bf16* wbase = weff + (size_t)b * C_OUT * KS * C_IN;
    #pragma unroll
    for (int s = 0; s < KS; ++s) {
      v16bf afrag[2];
      #pragma unroll
      for (int m = 0; m < 2; ++m) {
        const int c = waveC + m * 16 + lq;       // A row (M) per lane
        const __bf16* wp = wbase + ((size_t)c * KS + s) * C_IN + kc * 32 + half * 8;
        v8bf lo = *(const v8bf*)(wp);            // K = half*8 + 0..7
        v8bf hi = *(const v8bf*)(wp + 16);       // K = 16 + half*8 + 0..7
        afrag[m] = __builtin_shufflevector(lo, hi,
                     0,1,2,3,4,5,6,7,8,9,10,11,12,13,14,15);
      }
      #pragma unroll
      for (int n = 0; n < 4; ++n) {
        const int tl = wTloc + n * 16 + lq + s;  // shifted column per tap
        const __bf16* bp = xcur + tl * LROW + half * 8;
        v8bf b0 = *(const v8bf*)(bp);            // K = half*8 + 0..7
        v8bf b1 = *(const v8bf*)(bp + 16);       // K = 16 + half*8 + 0..7
        v16bf bfrag = __builtin_shufflevector(b0, b1,
                        0,1,2,3,4,5,6,7,8,9,10,11,12,13,14,15);
        acc[0][n] = __builtin_amdgcn_wmma_f32_16x16x32_bf16(
            false, afrag[0], false, bfrag, (short)0, acc[0][n], false, false);
        acc[1][n] = __builtin_amdgcn_wmma_f32_16x16x32_bf16(
            false, afrag[1], false, bfrag, (short)0, acc[1][n], false, false);
      }
    }

    if constexpr (MODE == 2) {
      if (kc < 3) {
        if (wave == 0) wait_tensorcnt0();
        __syncthreads();
      }
    } else {
      __syncthreads();
    }
  }

  // ---- epilogue: routed bias + store ----
  const float a0 = alpha[b*4+0], a1 = alpha[b*4+1];
  const float a2 = alpha[b*4+2], a3 = alpha[b*4+3];
  #pragma unroll
  for (int m = 0; m < 2; ++m) {
    float be[8];
    #pragma unroll
    for (int r = 0; r < 8; ++r) {
      const int c = waveC + m * 16 + half * 8 + r;
      be[r] = a0 * bias[0*C_OUT + c] + a1 * bias[1*C_OUT + c]
            + a2 * bias[2*C_OUT + c] + a3 * bias[3*C_OUT + c];
    }
    #pragma unroll
    for (int n = 0; n < 4; ++n) {
      const int t = waveT + n * 16 + lq;
      #pragma unroll
      for (int r = 0; r < 8; ++r) {
        const int c = waveC + m * 16 + half * 8 + r;
        out[((size_t)b * C_OUT + c) * T_LEN + t] = acc[m][n][r] + be[r];
      }
    }
  }
}

extern "C" void kernel_launch(void* const* d_in, const int* in_sizes, int n_in,
                              void* d_out, int out_size, void* d_ws, size_t ws_size,
                              hipStream_t stream) {
  const float* x      = (const float*)d_in[0];
  const float* weight = (const float*)d_in[1];
  const float* bias   = (const float*)d_in[2];
  const float* rw     = (const float*)d_in[3];
  const float* rb     = (const float*)d_in[4];
  float* out = (float*)d_out;

  char* ws = (char*)d_ws;
  float*  alpha  = (float*)(ws + WS_ALPHA);
  float*  pooled = (float*)(ws + WS_POOLED);
  __bf16* weff   = (__bf16*)(ws + WS_WEFF);
  __bf16* xbf    = (__bf16*)(ws + WS_XBF);

  pool_kernel <<<B_N * C_IN, 256, 0, stream>>>(x, pooled);
  route_kernel<<<1, 32, 0, stream>>>(pooled, rw, rb, alpha);
  weff_kernel <<<B_N * C_OUT, C_IN, 0, stream>>>(weight, alpha, weff);

  dim3 grid(T_LEN / 128, C_OUT / 128, B_N);
  constexpr int XMODE = HAVE_TDM ? 2 : 1;
  if (ws_size >= WS_XBF + XBF_BYTES) {
    xpad_kernel<<<256, 256, 0, stream>>>(xbf);
    xbf_kernel<<<dim3(T_LEN / 32, B_N), 256, 0, stream>>>(x, xbf);
    conv_wmma_kernel<XMODE><<<grid, 256, 0, stream>>>(x, xbf, weff, bias, alpha, out);
  } else {
    conv_wmma_kernel<0><<<grid, 256, 0, stream>>>(x, xbf, weff, bias, alpha, out);
  }
}
